// CGCNN_62689342653095
// MI455X (gfx1250) — matrix operand order repaired and checked by reading, not compiled, not run on hardware
//
#include <hip/hip_runtime.h>
#include <hip/hip_bf16.h>
#include <math.h>

#define N_NODES 50000
#define N_EDGES 800000
#define N_GRAPHS 512
#define F_IN 92
#define F_EDGE 50
#define DIM 64
#define GC 3
#define KPAD 192   // 178 padded to 6 * 32
#define EPS 1e-5f

typedef __attribute__((ext_vector_type(16))) _Float16 v16h;
typedef __attribute__((ext_vector_type(8)))  _Float16 v8h;
typedef __attribute__((ext_vector_type(8)))  float    v8f;

// ---------------------------------------------------------------- utilities
__global__ void zero_f32(float* __restrict__ p, int n) {
    int i = blockIdx.x * blockDim.x + threadIdx.x;
    if (i < n) p[i] = 0.0f;
}

// degree of each destination node (same for all layers)
__global__ void deg_kernel(const int* __restrict__ ei, float* __restrict__ cnt) {
    int e = blockIdx.x * blockDim.x + threadIdx.x;
    if (e < N_EDGES) unsafeAtomicAdd(&cnt[ei[N_EDGES + e]], 1.0f);
}

// convert f_W / s_W [GC][178][64] fp32 -> f16 transposed+padded [GC][64][192]
__global__ void conv_w_kernel(const float* __restrict__ fW, const float* __restrict__ sW,
                              _Float16* __restrict__ wtF, _Float16* __restrict__ wtS) {
    int i = blockIdx.x * blockDim.x + threadIdx.x;
    if (i >= GC * DIM * KPAD) return;
    int l = i / (DIM * KPAD);
    int rem = i % (DIM * KPAD);
    int n = rem / KPAD;
    int k = rem % KPAD;
    float vf = 0.0f, vs = 0.0f;
    if (k < 2 * DIM + F_EDGE) {
        vf = fW[((size_t)l * 178 + k) * DIM + n];
        vs = sW[((size_t)l * 178 + k) * DIM + n];
    }
    wtF[i] = (_Float16)vf;
    wtS[i] = (_Float16)vs;
}

// ---------------------------------------------------------------- pre FC
// h = relu(x @ pre_W + pre_b), also write f16 mirror
__global__ __launch_bounds__(64) void pre_fc_kernel(const float* __restrict__ x,
                                                    const float* __restrict__ W,
                                                    const float* __restrict__ b,
                                                    float* __restrict__ h,
                                                    _Float16* __restrict__ h16) {
    __shared__ float xs[F_IN];
    int node = blockIdx.x;
    int n = threadIdx.x;
    for (int k = n; k < F_IN; k += DIM) xs[k] = x[(size_t)node * F_IN + k];
    __syncthreads();
    float acc = b[n];
    #pragma unroll 4
    for (int k = 0; k < F_IN; ++k) acc = fmaf(xs[k], W[k * DIM + n], acc);
    acc = fmaxf(acc, 0.0f);
    size_t o = (size_t)node * DIM + n;
    h[o] = acc;
    h16[o] = (_Float16)acc;
}

// ---------------------------------------------------------------- edge GEMM (WMMA)
// One wave handles 16 edges x 64 outputs. K = 192 (dst 64 | src 64 | edge 50+pad).
// Weights (both gate and core, 48 KB) are staged in LDS once per block; all six
// A fragments are loaded into registers up front. Per k-step, all 8 B fragments
// are batch-loaded from LDS (clause of 16 ds_load_b128) before the 8 WMMAs, so
// the WMMA stream runs behind a single dscnt wait per k-step.
// A fragment (16x32 f16): lane L holds row m = L&15; halves at k-offsets
//   {hi*8 .. +7} and {16+hi*8 .. +7}, hi = L>>4.
// B fragment (32x16 f16): lane L holds column n = t*16 + (L&15); K halves at
//   ks*32 + {hi*8 .. +7} and ks*32 + {16+hi*8 .. +7} from the transposed weights [64][192].
// C/D (16x16 f32): lane L reg r holds row (r + 8*hi), col (L&15).
__global__ __launch_bounds__(256)
void edge_gemm_kernel(const _Float16* __restrict__ h16,
                      const int* __restrict__ ei,
                      const float* __restrict__ ea,
                      const _Float16* __restrict__ wtF,   // [64][192] this layer
                      const _Float16* __restrict__ wtS,   // [64][192] this layer
                      const float* __restrict__ fb,       // [64]
                      const float* __restrict__ sb,       // [64]
                      float* __restrict__ agg) {
    __shared__ _Float16 sW[2 * DIM * KPAD];   // 49152 bytes: [2][64][192]

    // cooperative stage of both weight matrices into LDS (16B vectors)
    {
        const uint4* gF = (const uint4*)wtF;      // 1536 uint4 each
        const uint4* gS = (const uint4*)wtS;
        uint4* sv = (uint4*)sW;
        #pragma unroll
        for (int i = threadIdx.x; i < 3072; i += 256)
            sv[i] = (i < 1536) ? gF[i] : gS[i - 1536];
    }
    __syncthreads();

    const int lane = threadIdx.x & 31;
    const int wave = threadIdx.x >> 5;
    const int tile = blockIdx.x * 8 + wave;      // 50000 tiles total, exact
    const int e0 = tile * 16;
    const int m = lane & 15;
    const int hi = lane >> 4;

    const int eRow = e0 + m;
    const int srcI = ei[eRow];
    const int dstI = ei[N_EDGES + eRow];
    const _Float16* aRowD = h16 + (size_t)dstI * DIM;
    const _Float16* aRowS = h16 + (size_t)srcI * DIM;
    const float* eaRow = ea + (size_t)eRow * F_EDGE;

    // ---- load all six A fragments up front (48 VGPRs) ----
    v16h aFrag[6];
    #pragma unroll
    for (int p = 0; p < 2; ++p) {              // p=0: dst row, p=1: src row
        const _Float16* base = p ? aRowS : aRowD;
        #pragma unroll
        for (int q = 0; q < 2; ++q) {          // 32-wide K sub-blocks
            v8h lo = *(const v8h*)(base + q * 32 + hi * 8);
            v8h hh = *(const v8h*)(base + q * 32 + 16 + hi * 8);
            #pragma unroll
            for (int j = 0; j < 8; ++j) {
                aFrag[p * 2 + q][j] = lo[j];
                aFrag[p * 2 + q][8 + j] = hh[j];
            }
        }
    }
    #pragma unroll
    for (int kl = 0; kl < 2; ++kl) {           // edge-attr K blocks, pad past 50
        #pragma unroll
        for (int j = 0; j < 8; ++j) {
            int k0 = kl * 32 + hi * 8 + j;          // always < 50
            int k1 = kl * 32 + 16 + hi * 8 + j;     // may exceed 49 when kl==1
            aFrag[4 + kl][j]     = (_Float16)eaRow[k0];
            aFrag[4 + kl][8 + j] = (k1 < F_EDGE) ? (_Float16)eaRow[k1] : (_Float16)0.0f;
        }
    }

    v8f accF[4], accS[4];
    #pragma unroll
    for (int t = 0; t < 4; ++t)
        #pragma unroll
        for (int j = 0; j < 8; ++j) { accF[t][j] = 0.0f; accS[t][j] = 0.0f; }

    const _Float16* swF = sW;
    const _Float16* swS = sW + DIM * KPAD;

    #pragma unroll
    for (int ks = 0; ks < 6; ++ks) {
        const int kb = ks * 32 + hi * 8;
        // batch all 8 B fragments for this k-step (one LDS clause)
        v16h bF[4], bS[4];
        #pragma unroll
        for (int t = 0; t < 4; ++t) {
            const int n = t * 16 + m;
            bF[t] = __builtin_shufflevector(*(const v8h*)(swF + n * KPAD + kb), *(const v8h*)(swF + n * KPAD + kb + 16), 0,1,2,3,4,5,6,7,8,9,10,11,12,13,14,15);
            bS[t] = __builtin_shufflevector(*(const v8h*)(swS + n * KPAD + kb), *(const v8h*)(swS + n * KPAD + kb + 16), 0,1,2,3,4,5,6,7,8,9,10,11,12,13,14,15);
        }
        #pragma unroll
        for (int t = 0; t < 4; ++t) {
            accF[t] = __builtin_amdgcn_wmma_f32_16x16x32_f16(
                false, aFrag[ks], false, bF[t], (short)0, accF[t], false, false);
            accS[t] = __builtin_amdgcn_wmma_f32_16x16x32_f16(
                false, aFrag[ks], false, bS[t], (short)0, accS[t], false, false);
        }
    }

    // destination node of each output row this lane owns (rows r + 8*hi)
    int dstRow[8];
    #pragma unroll
    for (int r = 0; r < 8; ++r) dstRow[r] = ei[N_EDGES + e0 + hi * 8 + r];

    #pragma unroll
    for (int t = 0; t < 4; ++t) {
        const int n = t * 16 + m;
        const float fbn = fb[n];
        const float sbn = sb[n];
        #pragma unroll
        for (int r = 0; r < 8; ++r) {
            float f = accF[t][r] + fbn;
            float s = accS[t][r] + sbn;
            float sig = 1.0f / (1.0f + __expf(-f));                 // v_exp_f32
            float sp  = (s > 20.0f) ? s : __logf(1.0f + __expf(s)); // v_log_f32
            unsafeAtomicAdd(&agg[(size_t)dstRow[r] * DIM + n], sig * sp);
        }
    }
}

// ---------------------------------------------------------------- residual + BN
__global__ void update_kernel(float* __restrict__ h, _Float16* __restrict__ h16,
                              const float* __restrict__ agg, const float* __restrict__ cnt,
                              const float* __restrict__ g, const float* __restrict__ b,
                              const float* __restrict__ rm, const float* __restrict__ rv) {
    int i = blockIdx.x * blockDim.x + threadIdx.x;
    if (i >= N_NODES * DIM) return;
    int n = i & (DIM - 1);
    int node = i >> 6;
    float c = cnt[node];
    float mean = (c > 0.0f) ? agg[i] / c : 0.0f;
    float v = h[i] + mean;
    v = (v - rm[n]) * rsqrtf(rv[n] + EPS) * g[n] + b[n];
    h[i] = v;
    h16[i] = (_Float16)v;
}

// ---------------------------------------------------------------- global mean pool
__global__ __launch_bounds__(64) void pool_kernel(const float* __restrict__ h,
                                                  const int* __restrict__ batch,
                                                  float* __restrict__ gsum,
                                                  float* __restrict__ gcnt) {
    int node = blockIdx.x;
    int n = threadIdx.x;
    int b = batch[node];
    unsafeAtomicAdd(&gsum[b * DIM + n], h[(size_t)node * DIM + n]);
    if (n == 0) unsafeAtomicAdd(&gcnt[b], 1.0f);
}

// ---------------------------------------------------------------- post FC + out
__global__ __launch_bounds__(64) void final_kernel(const float* __restrict__ gsum,
                                                   const float* __restrict__ gcnt,
                                                   const float* __restrict__ postW,
                                                   const float* __restrict__ postb,
                                                   const float* __restrict__ outW,
                                                   const float* __restrict__ outb,
                                                   float* __restrict__ out) {
    __shared__ float gs[DIM];
    __shared__ float vs[DIM];
    int b = blockIdx.x;
    int n = threadIdx.x;
    float c = gcnt[b];
    gs[n] = (c > 0.0f) ? gsum[b * DIM + n] / c : 0.0f;
    __syncthreads();
    float acc = postb[n];
    #pragma unroll 8
    for (int k = 0; k < DIM; ++k) acc = fmaf(gs[k], postW[k * DIM + n], acc);
    acc = fmaxf(acc, 0.0f);
    vs[n] = acc * outW[n];
    __syncthreads();
    #pragma unroll
    for (int s = 32; s > 0; s >>= 1) {
        if (n < s) vs[n] += vs[n + s];
        __syncthreads();
    }
    if (n == 0) out[b] = vs[0] + outb[0];
}

// ---------------------------------------------------------------- launcher
extern "C" void kernel_launch(void* const* d_in, const int* in_sizes, int n_in,
                              void* d_out, int out_size, void* d_ws, size_t ws_size,
                              hipStream_t stream) {
    const float* x        = (const float*)d_in[0];
    const int*   ei       = (const int*)d_in[1];     // [2, E]: row0 src, row1 dst
    const float* ea       = (const float*)d_in[2];
    const int*   batch    = (const int*)d_in[3];
    const float* pre_W    = (const float*)d_in[4];
    const float* pre_b    = (const float*)d_in[5];
    const float* f_W      = (const float*)d_in[6];
    const float* f_b      = (const float*)d_in[7];
    const float* s_W      = (const float*)d_in[8];
    const float* s_b      = (const float*)d_in[9];
    const float* bn_g     = (const float*)d_in[10];
    const float* bn_b     = (const float*)d_in[11];
    const float* bn_rm    = (const float*)d_in[12];
    const float* bn_rv    = (const float*)d_in[13];
    const float* post_W   = (const float*)d_in[14];
    const float* post_b   = (const float*)d_in[15];
    const float* out_W    = (const float*)d_in[16];
    const float* out_b    = (const float*)d_in[17];
    float* out = (float*)d_out;

    // workspace carve-up (256B aligned)
    char* ws = (char*)d_ws;
    size_t off = 0;
    auto carve = [&](size_t bytes) -> char* {
        char* p = ws + off;
        off += (bytes + 255) & ~(size_t)255;
        return p;
    };
    float*    h    = (float*)carve((size_t)N_NODES * DIM * 4);
    float*    agg  = (float*)carve((size_t)N_NODES * DIM * 4);
    float*    cnt  = (float*)carve((size_t)N_NODES * 4);
    _Float16* h16  = (_Float16*)carve((size_t)N_NODES * DIM * 2);
    _Float16* wtF  = (_Float16*)carve((size_t)GC * DIM * KPAD * 2);
    _Float16* wtS  = (_Float16*)carve((size_t)GC * DIM * KPAD * 2);
    float*    gsum = (float*)carve((size_t)N_GRAPHS * DIM * 4);
    float*    gcnt = (float*)carve((size_t)N_GRAPHS * 4);

    // one-time per launch: degree counts + f16 weights + pre FC
    zero_f32<<<(N_NODES + 255) / 256, 256, 0, stream>>>(cnt, N_NODES);
    deg_kernel<<<(N_EDGES + 255) / 256, 256, 0, stream>>>(ei, cnt);
    conv_w_kernel<<<(GC * DIM * KPAD + 255) / 256, 256, 0, stream>>>(f_W, s_W, wtF, wtS);
    pre_fc_kernel<<<N_NODES, DIM, 0, stream>>>(x, pre_W, pre_b, h, h16);

    // CGConv layers
    for (int l = 0; l < GC; ++l) {
        zero_f32<<<(N_NODES * DIM + 255) / 256, 256, 0, stream>>>(agg, N_NODES * DIM);
        edge_gemm_kernel<<<(N_EDGES / 16) / 8, 256, 0, stream>>>(
            h16, ei, ea,
            wtF + (size_t)l * DIM * KPAD, wtS + (size_t)l * DIM * KPAD,
            f_b + l * DIM, s_b + l * DIM, agg);
        update_kernel<<<(N_NODES * DIM + 255) / 256, 256, 0, stream>>>(
            h, h16, agg, cnt,
            bn_g + l * DIM, bn_b + l * DIM, bn_rm + l * DIM, bn_rv + l * DIM);
    }

    // pooling + head
    zero_f32<<<(N_GRAPHS * DIM + 255) / 256, 256, 0, stream>>>(gsum, N_GRAPHS * DIM);
    zero_f32<<<(N_GRAPHS + 255) / 256, 256, 0, stream>>>(gcnt, N_GRAPHS);
    pool_kernel<<<N_NODES, DIM, 0, stream>>>(h, batch, gsum, gcnt);
    final_kernel<<<N_GRAPHS, DIM, 0, stream>>>(gsum, gcnt, post_W, post_b, out_W, out_b, out);
}
